// LSTMAutoencoder_38173669326873
// MI455X (gfx1250) — compile-verified, matrix operand order checked by reading, not run on hardware
//
#include <hip/hip_runtime.h>
#include <stdint.h>

// Problem constants
#define B_    64
#define T_    256
#define D_    409
#define H_    1024
#define G4H   4096
#define DPADX 512            // x padded to 512 so all K-chunks are uniform
#define KTOT  1536           // H + DPADX (fused [h | x] K dimension)
#define KCHUNK 256           // K staged per TDM chunk (8 WMMA k-steps)
#define NCHUNKS 6            // chunks 0..3 from h, 4..5 from x
#define KSTEPS (KCHUNK / 32)
#define LDSTRIDE 264         // bf16 elems per LDS row = 512B tile row + 16B TDM pad

typedef __attribute__((ext_vector_type(16))) __bf16 v16bf;
typedef __attribute__((ext_vector_type(8)))  float  v8f;
typedef unsigned int u32x4 __attribute__((ext_vector_type(4)));
typedef int          i32x4 __attribute__((ext_vector_type(4)));
typedef int          i32x8 __attribute__((ext_vector_type(8)));

union FragAB { v16bf v; uint4 q[2]; };

__device__ __forceinline__ unsigned short f2bf(float f) {
  unsigned u = __builtin_bit_cast(unsigned, f);
  u += 0x7FFFu + ((u >> 16) & 1u);   // round-to-nearest-even
  return (unsigned short)(u >> 16);
}
__device__ __forceinline__ float sigmoidf_(float x) {
  return 1.f / (1.f + __expf(-x));
}
__device__ __forceinline__ float tanhf_(float x) {
  float a = fabsf(x);
  float e = __expf(-2.f * a);
  float t = (1.f - e) / (1.f + e);
  return copysignf(t, x);
}

// Issue one 2D TDM tile load: tile_h rows x KCHUNK bf16 elems from global
// (row stride = stride_elems) into LDS at lds_off, with 4-DWORD padding after
// every 128 DWORDs (one 512B tile row) -> LDS row stride = 528B, bank-friendly.
__device__ __forceinline__ void tdm_load_chunk(unsigned lds_off,
                                               const unsigned short* gsrc,
                                               unsigned stride_elems) {
  unsigned long long ga = (unsigned long long)(uintptr_t)gsrc;
  u32x4 g0;
  g0[0] = 1u;                                   // count=1, user descriptor
  g0[1] = lds_off;                              // lds_addr (bytes)
  g0[2] = (unsigned)(ga & 0xFFFFFFFFu);         // global_addr[31:0]
  g0[3] = (unsigned)((ga >> 32) & 0x1FFFFFFu)   // global_addr[56:32]
          | (2u << 30);                         // type = 2 ("image")
  i32x8 g1;
  // data_size=1 (2B), pad_enable, pad_interval code 6 (=128 DWORDs),
  // pad_amount code 3 (=4 DWORDs)
  g1[0] = (int)((1u << 16) | (1u << 20) | (6u << 22) | (3u << 25));
  // tensor_dim0 = tensor_dim1 = 16M (tile never OOB), laid per bit map:
  g1[1] = 0;                                    // abar=0 | dim0[15:0]=0
  g1[2] = (int)0x0100u;                         // dim1[15:0]=0 | dim0[31:16]
  g1[3] = (int)(((unsigned)KCHUNK << 16) | 0x0100u); // tile_dim0 | dim1[31:16]
  g1[4] = (int)(unsigned)B_;                    // tile_dim2=0 | tile_dim1=64
  g1[5] = (int)stride_elems;                    // tensor_dim0_stride[31:0]
  g1[6] = 0;                                    // dim1_stride lo | stride[47:32]
  g1[7] = 0;
  i32x4 gz = {0, 0, 0, 0};
#if __clang_major__ >= 23
  i32x8 gz8 = {0, 0, 0, 0, 0, 0, 0, 0};
  __builtin_amdgcn_tensor_load_to_lds(g0, g1, gz, gz, gz8, 0);
#else
  __builtin_amdgcn_tensor_load_to_lds(g0, g1, gz, gz, 0);
#endif
}

// ---------------- prep kernels ----------------

// inputs [B,T,D] f32 -> time-major bf16 [T][B][DPADX]
__global__ void k_convert_x(const float* __restrict__ inp,
                            unsigned short* __restrict__ xbf) {
  int idx = blockIdx.x * blockDim.x + threadIdx.x;
  const int N = T_ * B_ * DPADX;
  if (idx >= N) return;
  int k = idx % DPADX;
  int b = (idx / DPADX) % B_;
  int t = idx / (DPADX * B_);
  float v = (k < D_) ? inp[((size_t)b * T_ + t) * D_ + k] : 0.f;
  xbf[idx] = f2bf(v);
}

// W[n][k] = k<H ? Whh[n][k] : (k<H+D ? Wih[n][k-H] : 0), bf16, [4096][1536]
__global__ void k_build_wcomb(const float* __restrict__ Whh,
                              const float* __restrict__ Wih,
                              unsigned short* __restrict__ W) {
  int idx = blockIdx.x * blockDim.x + threadIdx.x;
  const int N = G4H * KTOT;
  if (idx >= N) return;
  int k = idx % KTOT;
  int n = idx / KTOT;
  float v;
  if (k < H_)              v = Whh[(size_t)n * H_ + k];
  else if (k < H_ + D_)    v = Wih[(size_t)n * D_ + (k - H_)];
  else                     v = 0.f;
  W[idx] = f2bf(v);
}

// pred_W [409][1024] f32 -> padded bf16 [416][1024]
__global__ void k_build_predw(const float* __restrict__ pW,
                              unsigned short* __restrict__ out) {
  int idx = blockIdx.x * blockDim.x + threadIdx.x;
  const int N = 416 * H_;
  if (idx >= N) return;
  int n = idx / H_;
  int k = idx % H_;
  out[idx] = f2bf(n < D_ ? pW[(size_t)n * H_ + k] : 0.f);
}

// zero h double-buffer (as u32), c (f32), barrier counters
__global__ void k_init(unsigned* __restrict__ hbuf32,
                       float* __restrict__ c,
                       unsigned* __restrict__ cnt) {
  int idx = blockIdx.x * blockDim.x + threadIdx.x;
  if (idx < 65536) { hbuf32[idx] = 0u; c[idx] = 0.f; }
  if (idx < 4) cnt[idx] = 0u;
}

// ---------------- persistent recurrent LSTM ----------------
// 16 blocks x 128 threads (4 waves). Wave owns j-tile jt = blk*4 + wave
// (16 columns of H), all 64 batch rows, all 4 gates.
template <bool IS_ENC>
__global__ __launch_bounds__(128) void k_lstm(
    const unsigned short* __restrict__ xbf,   // [T][64][512] bf16
    const unsigned short* __restrict__ W,     // [4096][1536] bf16
    const float* __restrict__ bias,           // [4096] f32 (gate-major i,f,g,o)
    unsigned short* __restrict__ hbuf,        // [2][64][1024] bf16
    float* __restrict__ c,                    // [64][1024] f32 (persistent enc->dec)
    unsigned short* __restrict__ hs,          // [T][64][1024] bf16 (decoder only)
    float* __restrict__ latents,              // [64][1024] f32 (encoder only)
    unsigned* __restrict__ cnt)               // grid barrier counter
{
  __shared__ __align__(16) unsigned short lds[B_ * LDSTRIDE];

  const int tid   = threadIdx.x;
  const int lane  = tid & 31;
  const int wave  = tid >> 5;
  const int jt    = blockIdx.x * 4 + wave;   // 0..63
  const int nlo   = lane & 15;               // M (A) / N (B) index within tile
  const int khalf = lane >> 4;               // K half selector
  const int j     = jt * 16 + nlo;           // hidden column this lane produces
  const unsigned ldsbase = (unsigned)(uintptr_t)(void*)&lds[0];

  const float b0 = bias[0 * H_ + j];
  const float b1 = bias[1 * H_ + j];
  const float b2 = bias[2 * H_ + j];
  const float b3 = bias[3 * H_ + j];

  volatile unsigned* vcnt = cnt;

  for (int t = 0; t < T_; ++t) {
    const unsigned short* hprev = hbuf + (size_t)(t & 1) * B_ * H_;
    unsigned short*       hnext = hbuf + (size_t)((t + 1) & 1) * B_ * H_;
    const int xt = IS_ENC ? t : (t - 1);     // decoder input is shifted x

    v8f acc[4][4];
#pragma unroll
    for (int g = 0; g < 4; ++g)
#pragma unroll
      for (int mt = 0; mt < 4; ++mt)
        acc[g][mt] = (v8f){0.f, 0.f, 0.f, 0.f, 0.f, 0.f, 0.f, 0.f};

#pragma unroll 1
    for (int ch = 0; ch < NCHUNKS; ++ch) {
      const int kc = ch * KCHUNK;
      const bool fromH = (kc < H_);
      __syncthreads();   // previous chunk's LDS reads consumed (frag data used)
      if (!fromH && xt < 0) {
        // decoder t==0: x is zeros -> zero-fill LDS chunk
        for (int idx = tid; idx < B_ * (LDSTRIDE / 8); idx += 128)
          *(uint4*)(lds + idx * 8) = uint4{0u, 0u, 0u, 0u};
      } else if (wave == 0) {
        const unsigned short* src =
            fromH ? (hprev + kc)
                  : (xbf + (size_t)xt * B_ * DPADX + (kc - H_));
        tdm_load_chunk(ldsbase, src, fromH ? H_ : DPADX);
        __builtin_amdgcn_s_wait_tensorcnt((short)0);
      }
      __syncthreads();

#pragma unroll 1
      for (int ks = 0; ks < KSTEPS; ++ks) {
        const int k0 = ks * 32;
        // A fragments (16x32 bf16) for 4 batch tiles, from LDS
        FragAB A[4];
#pragma unroll
        for (int mt = 0; mt < 4; ++mt) {
          const unsigned short* ap =
              lds + (mt * 16 + nlo) * LDSTRIDE + k0 + khalf * 8;
          A[mt].q[0] = *(const uint4*)(ap);
          A[mt].q[1] = *(const uint4*)(ap + 16);
        }
#pragma unroll
        for (int g = 0; g < 4; ++g) {
          // B fragment (32x16 bf16): lane n reads row (gate*H + jt*16 + n) of W
          FragAB Bf;
          const unsigned short* bp =
              W + (size_t)(g * H_ + jt * 16 + nlo) * KTOT + kc + k0 + khalf * 8;
          Bf.q[0] = *(const uint4*)(bp);
          Bf.q[1] = *(const uint4*)(bp + 16);
#pragma unroll
          for (int mt = 0; mt < 4; ++mt) {
            acc[g][mt] = __builtin_amdgcn_wmma_f32_16x16x32_bf16(
                false, A[mt].v, false, Bf.v, (short)0, acc[g][mt], false, false);
          }
        }
      }
    }

    // Activations + state update. Lane owns column j, rows mt*16+khalf*8+r.
#pragma unroll
    for (int mt = 0; mt < 4; ++mt) {
#pragma unroll
      for (int r = 0; r < 8; ++r) {
        int m = mt * 16 + khalf * 8 + r;
        size_t cidx = (size_t)m * H_ + j;
        float ig = sigmoidf_(acc[0][mt][r] + b0);
        float fg = sigmoidf_(acc[1][mt][r] + b1);
        float gg = tanhf_(acc[2][mt][r] + b2);
        float og = sigmoidf_(acc[3][mt][r] + b3);
        float cn = fg * c[cidx] + ig * gg;
        c[cidx] = cn;
        float hv = og * tanhf_(cn);
        hnext[cidx] = f2bf(hv);
        if (!IS_ENC) hs[((size_t)t * B_ + m) * H_ + j] = f2bf(hv);
        if (IS_ENC && t == T_ - 1) latents[cidx] = hv;
      }
    }

    // Grid-wide step barrier (monotonic counter; zeroed each launch by k_init)
    __threadfence();
    __syncthreads();
    if (tid == 0) {
      atomicAdd(cnt, 1u);
      unsigned target = (unsigned)(t + 1) * gridDim.x;
      while (*vcnt < target) __builtin_amdgcn_s_sleep(1);
    }
    __syncthreads();
    __threadfence();
  }
}

// ---------------- prediction head ----------------
// seq[b][t][n] = sum_k hs[t][b][k] * pred_W[n][k] + pred_b[n]
__global__ __launch_bounds__(128) void k_head(
    const unsigned short* __restrict__ hs,  // [16384][1024] bf16 (t-major)
    const unsigned short* __restrict__ Wp,  // [416][1024] bf16
    const float* __restrict__ pb,           // [409] f32
    float* __restrict__ out)                // seq base ([B][T][D] f32)
{
  const int lane  = threadIdx.x & 31;
  const int wave  = threadIdx.x >> 5;
  const int tile  = blockIdx.x * 4 + wave;
  const int NT    = 416 / 16;               // 26 n-tiles
  const int MT    = (T_ * B_) / 16;         // 1024 m-tiles
  if (tile >= MT * NT) return;
  const int mt = tile / NT;
  const int nt = tile % NT;
  const int nlo = lane & 15;
  const int khalf = lane >> 4;

  v8f acc = (v8f){0.f, 0.f, 0.f, 0.f, 0.f, 0.f, 0.f, 0.f};
  const unsigned short* arow = hs + (size_t)(mt * 16 + nlo) * H_ + khalf * 8;
  const unsigned short* brow = Wp + (size_t)(nt * 16 + nlo) * H_ + khalf * 8;
#pragma unroll 4
  for (int k0 = 0; k0 < H_; k0 += 32) {
    FragAB A, Bf;
    A.q[0]  = *(const uint4*)(arow + k0);
    A.q[1]  = *(const uint4*)(arow + k0 + 16);
    Bf.q[0] = *(const uint4*)(brow + k0);
    Bf.q[1] = *(const uint4*)(brow + k0 + 16);
    acc = __builtin_amdgcn_wmma_f32_16x16x32_bf16(
        false, A.v, false, Bf.v, (short)0, acc, false, false);
  }

  int n = nt * 16 + nlo;
  if (n < D_) {
    float bias = pb[n];
#pragma unroll
    for (int r = 0; r < 8; ++r) {
      int row = mt * 16 + khalf * 8 + r;   // row = t*64 + b
      int b = row & (B_ - 1);
      int t = row >> 6;
      out[((size_t)b * T_ + t) * D_ + n] = acc[r] + bias;
    }
  }
}

// ---------------- host launch ----------------
extern "C" void kernel_launch(void* const* d_in, const int* in_sizes, int n_in,
                              void* d_out, int out_size, void* d_ws, size_t ws_size,
                              hipStream_t stream) {
  const float* inputs  = (const float*)d_in[0];
  const float* enc_Wih = (const float*)d_in[1];
  const float* enc_Whh = (const float*)d_in[2];
  const float* enc_b   = (const float*)d_in[3];
  const float* dec_Wih = (const float*)d_in[4];
  const float* dec_Whh = (const float*)d_in[5];
  const float* dec_b   = (const float*)d_in[6];
  const float* pred_W  = (const float*)d_in[7];
  const float* pred_b  = (const float*)d_in[8];
  float* out = (float*)d_out;

  // Workspace layout (all 256B-aligned)
  char* ws = (char*)d_ws;
  const size_t OFF_XBF   = 0;                                   // 16,777,216
  const size_t OFF_WENC  = OFF_XBF  + (size_t)T_ * B_ * DPADX * 2;
  const size_t OFF_WDEC  = OFF_WENC + (size_t)G4H * KTOT * 2;   // 12,582,912 each
  const size_t OFF_PREDW = OFF_WDEC + (size_t)G4H * KTOT * 2;
  const size_t OFF_HBUF  = OFF_PREDW + (size_t)416 * H_ * 2;    //    851,968
  const size_t OFF_C     = OFF_HBUF + (size_t)2 * B_ * H_ * 2;  //    262,144
  const size_t OFF_HS    = OFF_C    + (size_t)B_ * H_ * 4;      //    262,144
  const size_t OFF_CNT   = OFF_HS   + (size_t)T_ * B_ * H_ * 2; // 33,554,432
  const size_t NEEDED    = OFF_CNT + 256;
  if (ws_size < NEEDED) return;  // deterministic no-op if workspace too small

  unsigned short* xbf   = (unsigned short*)(ws + OFF_XBF);
  unsigned short* wenc  = (unsigned short*)(ws + OFF_WENC);
  unsigned short* wdec  = (unsigned short*)(ws + OFF_WDEC);
  unsigned short* predw = (unsigned short*)(ws + OFF_PREDW);
  unsigned short* hbuf  = (unsigned short*)(ws + OFF_HBUF);
  float*          cbuf  = (float*)(ws + OFF_C);
  unsigned short* hsbuf = (unsigned short*)(ws + OFF_HS);
  unsigned*       cnt   = (unsigned*)(ws + OFF_CNT);

  // prep
  {
    int N = T_ * B_ * DPADX;
    k_convert_x<<<(N + 255) / 256, 256, 0, stream>>>(inputs, xbf);
  }
  {
    int N = G4H * KTOT;
    k_build_wcomb<<<(N + 255) / 256, 256, 0, stream>>>(enc_Whh, enc_Wih, wenc);
    k_build_wcomb<<<(N + 255) / 256, 256, 0, stream>>>(dec_Whh, dec_Wih, wdec);
  }
  {
    int N = 416 * H_;
    k_build_predw<<<(N + 255) / 256, 256, 0, stream>>>(pred_W, predw);
  }
  k_init<<<65536 / 256, 256, 0, stream>>>((unsigned*)hbuf, cbuf, cnt);

  // encoder: h0=c0=0; writes latents (f32) on final step; final h in hbuf[0], c in place
  k_lstm<true><<<16, 128, 0, stream>>>(xbf, wenc, enc_b, hbuf, cbuf,
                                       hsbuf, out, cnt + 0);
  // decoder: continues from (hT, cT); records hs for the head
  k_lstm<false><<<16, 128, 0, stream>>>(xbf, wdec, dec_b, hbuf, cbuf,
                                        hsbuf, out, cnt + 1);

  // head: seq output after the 64*1024 latents
  {
    int tiles = ((T_ * B_) / 16) * (416 / 16);   // 26624
    k_head<<<tiles / 4, 128, 0, stream>>>(hsbuf, predw, pred_b,
                                          out + (size_t)B_ * H_);
  }
}
